// KnowledgeSelector_31834297598024
// MI455X (gfx1250) — matrix operand order repaired and checked by reading, not compiled, not run on hardware
//
#include <hip/hip_runtime.h>

// Problem constants (B, H, K from the reference)
#define BB   64
#define HH   2048
#define KK   256
#define RR   (KK * BB)          // 16384 flattened (k,b) rows
#define EPSV 1e-8f

typedef __bf16 bf16_t;
typedef bf16_t v16bf __attribute__((ext_vector_type(16)));
typedef bf16_t v8bf  __attribute__((ext_vector_type(8)));
typedef float  v8f   __attribute__((ext_vector_type(8)));

// ---------------------------------------------------------------------------
// Convert f32 -> bf16, 8 elements per thread (2x float4 loads, one 16B store)
// ---------------------------------------------------------------------------
__global__ __launch_bounds__(256) void cvt_bf16_kernel(const float* __restrict__ src,
                                                       bf16_t* __restrict__ dst) {
    int i = blockIdx.x * blockDim.x + threadIdx.x;      // i indexes groups of 8
    const float4* s = (const float4*)src + (size_t)i * 2;
    float4 x0 = s[0], x1 = s[1];
    v8bf o;
    o[0] = (bf16_t)x0.x; o[1] = (bf16_t)x0.y; o[2] = (bf16_t)x0.z; o[3] = (bf16_t)x0.w;
    o[4] = (bf16_t)x1.x; o[5] = (bf16_t)x1.y; o[6] = (bf16_t)x1.z; o[7] = (bf16_t)x1.w;
    *((v8bf*)dst + i) = o;
}

// ---------------------------------------------------------------------------
// A-fragment (16x32 bf16, M x K): lane m = lane&15, hi = lane>>4.
// Contiguous layout (matches B): VGPR0-3 hold K = hi*16 + 0..7, VGPR4-7 hold K = hi*16 + 8..15.
// Source rows are f32; convert in-register (v_cvt_pk_bf16_f32).
// ---------------------------------------------------------------------------
__device__ __forceinline__ v16bf load_a_frag(const float* __restrict__ rowPtr, int h0, int hi) {
    const float4* p0 = (const float4*)(rowPtr + h0 + hi * 16);
    const float4* p1 = (const float4*)(rowPtr + h0 + hi * 16 + 8);
    float4 a0 = p0[0], a1 = p0[1];
    float4 a2 = p1[0], a3 = p1[1];
    v16bf a;
    a[0]  = (bf16_t)a0.x; a[1]  = (bf16_t)a0.y; a[2]  = (bf16_t)a0.z; a[3]  = (bf16_t)a0.w;
    a[4]  = (bf16_t)a1.x; a[5]  = (bf16_t)a1.y; a[6]  = (bf16_t)a1.z; a[7]  = (bf16_t)a1.w;
    a[8]  = (bf16_t)a2.x; a[9]  = (bf16_t)a2.y; a[10] = (bf16_t)a2.z; a[11] = (bf16_t)a2.w;
    a[12] = (bf16_t)a3.x; a[13] = (bf16_t)a3.y; a[14] = (bf16_t)a3.z; a[15] = (bf16_t)a3.w;
    return a;
}

// ---------------------------------------------------------------------------
// Fused projection GEMM: D[r, o] = sum_h A[r, h] * W[o, h]  (+ bias[o])
//   STORE_Q = 1 : A = input_state (64 rows), store q = D + bias
//   STORE_Q = 0 : A = knowledge_base (16384 rows); do NOT store D; instead
//                 accumulate per-row  sum(D^2)  -> kn2[r]   (atomic f32)
//                              and    sum(D*q)  -> dots[r]  (atomic f32)
// Tile: 64 rows x 128 cols per WG, 8 waves as 4(M) x 2(N), 16x64 per wave,
// K-chunks of 32 with v_wmma_f32_16x16x32_bf16. All 4 B-fragments are loaded
// up front each chunk so the load clause drains under the matrix pipe.
// ---------------------------------------------------------------------------
template <int STORE_Q>
__global__ __launch_bounds__(256) void proj_kernel(const float*  __restrict__ Arows,
                                                   const bf16_t* __restrict__ Wb,
                                                   const float*  __restrict__ bias,
                                                   const float*  __restrict__ q,
                                                   float* __restrict__ qout,
                                                   float* __restrict__ kn2,
                                                   float* __restrict__ dots) {
    const int tid  = threadIdx.x;
    const int lane = tid & 31;
    const int w    = tid >> 5;
    const int wm   = w & 3;          // M-tile of this wave (0..3)
    const int wn   = w >> 2;         // N-half of this wave (0..1)
    const int n    = lane & 15;
    const int hi   = lane >> 4;

    const int rBase = blockIdx.x * 64 + wm * 16;
    const int oBase = blockIdx.y * 128 + wn * 64;

    const int aRowIdx = rBase + n;                          // A row this lane streams
    const float* aRow = Arows + (size_t)aRowIdx * HH;

    // Per-lane B row pointers (column o of the 32x16 fragment = row o of Wb)
    const bf16_t* bRow0 = Wb + (size_t)(oBase +  0 + n) * HH + hi * 16;
    const bf16_t* bRow1 = Wb + (size_t)(oBase + 16 + n) * HH + hi * 16;
    const bf16_t* bRow2 = Wb + (size_t)(oBase + 32 + n) * HH + hi * 16;
    const bf16_t* bRow3 = Wb + (size_t)(oBase + 48 + n) * HH + hi * 16;

    v8f acc[4] = {};

    for (int h0 = 0; h0 < HH; h0 += 32) {
        // Speculative prefetch of next A chunk (no guard: branch-free loop;
        // OOB prefetches are dropped silently).
        __builtin_prefetch(aRow + h0 + 32, 0, 1);           // global_prefetch_b8

        // Issue all B-fragment loads first (8x b128 in one clause), then A.
        v16bf b0 = *(const v16bf*)(bRow0 + h0);
        v16bf b1 = *(const v16bf*)(bRow1 + h0);
        v16bf b2 = *(const v16bf*)(bRow2 + h0);
        v16bf b3 = *(const v16bf*)(bRow3 + h0);
        v16bf af = load_a_frag(aRow, h0, hi);

        acc[0] = __builtin_amdgcn_wmma_f32_16x16x32_bf16(false, af, false, b0,
                                                         (short)0, acc[0], false, false);
        acc[1] = __builtin_amdgcn_wmma_f32_16x16x32_bf16(false, af, false, b1,
                                                         (short)0, acc[1], false, false);
        acc[2] = __builtin_amdgcn_wmma_f32_16x16x32_bf16(false, af, false, b2,
                                                         (short)0, acc[2], false, false);
        acc[3] = __builtin_amdgcn_wmma_f32_16x16x32_bf16(false, af, false, b3,
                                                         (short)0, acc[3], false, false);
    }

    // Epilogue. C/D layout: lane value v of acc = element (M = v + hi*8, N = n).
    float biasv[4];
#pragma unroll
    for (int nt = 0; nt < 4; ++nt) biasv[nt] = bias[oBase + nt * 16 + n];

#pragma unroll
    for (int v = 0; v < 8; ++v) {
        const int m    = v + hi * 8;
        const int orow = rBase + m;
        if (STORE_Q) {
#pragma unroll
            for (int nt = 0; nt < 4; ++nt)
                qout[(size_t)orow * HH + oBase + nt * 16 + n] = acc[nt][v] + biasv[nt];
        } else {
            const int brow = orow & (BB - 1);               // b = flat row % 64
            float s1 = 0.f, s2 = 0.f;
#pragma unroll
            for (int nt = 0; nt < 4; ++nt) {
                float d  = acc[nt][v] + biasv[nt];
                float qv = q[(size_t)brow * HH + oBase + nt * 16 + n];
                s1 = fmaf(d, d, s1);
                s2 = fmaf(d, qv, s2);
            }
            // Reduce across the 16 N-lanes of this half-wave.
#pragma unroll
            for (int off = 8; off >= 1; off >>= 1) {
                s1 += __shfl_xor(s1, off, 16);
                s2 += __shfl_xor(s2, off, 16);
            }
            if (n == 0) {
                atomicAdd(&kn2[orow],  s1);
                atomicAdd(&dots[orow], s2);
            }
        }
    }
}

// ---------------------------------------------------------------------------
// Per-b: qn2, cosine scores over K=256 items, softmax over K -> attn[K,B]
// ---------------------------------------------------------------------------
__global__ __launch_bounds__(256) void softmax_kernel(const float* __restrict__ q,
                                                      const float* __restrict__ dots,
                                                      const float* __restrict__ kn2,
                                                      float* __restrict__ attn) {
    const int b = blockIdx.x;
    const int t = threadIdx.x;           // t == k (K == 256 == blockDim)
    __shared__ float red[256];

    // ||q_b||^2
    float s = 0.f;
    for (int h = t; h < HH; h += 256) {
        float v = q[(size_t)b * HH + h];
        s = fmaf(v, v, s);
    }
    red[t] = s; __syncthreads();
    for (int st = 128; st > 0; st >>= 1) { if (t < st) red[t] += red[t + st]; __syncthreads(); }
    const float qn = fmaxf(sqrtf(red[0]), EPSV);
    __syncthreads();

    const float kn    = fmaxf(sqrtf(kn2[t * BB + b]), EPSV);
    const float score = dots[t * BB + b] / (qn * kn);

    red[t] = score; __syncthreads();
    for (int st = 128; st > 0; st >>= 1) { if (t < st) red[t] = fmaxf(red[t], red[t + st]); __syncthreads(); }
    const float mx = red[0];
    __syncthreads();

    const float e = expf(score - mx);
    red[t] = e; __syncthreads();
    for (int st = 128; st > 0; st >>= 1) { if (t < st) red[t] += red[t + st]; __syncthreads(); }
    attn[t * BB + b] = e / red[0];
}

// ---------------------------------------------------------------------------
// out[b,h] = input[b,h] + sum_k attn[k,b] * KB[k,b,h]   (KB pass 2, L2-resident)
// grid (H/1024, B); one float4 per thread; attn column staged in LDS.
// ---------------------------------------------------------------------------
__global__ __launch_bounds__(256) void output_kernel(const float* __restrict__ inp,
                                                     const float* __restrict__ KB,
                                                     const float* __restrict__ attn,
                                                     float* __restrict__ out) {
    const int b = blockIdx.y;
    __shared__ float a_s[KK];
    a_s[threadIdx.x] = attn[threadIdx.x * BB + b];
    __syncthreads();

    const int h4 = (blockIdx.x * 256 + threadIdx.x) * 4;
    float4 acc = make_float4(0.f, 0.f, 0.f, 0.f);
#pragma unroll 4
    for (int k = 0; k < KK; ++k) {
        const float4 v = *(const float4*)(KB + ((size_t)(k * BB + b)) * HH + h4);
        const float a = a_s[k];
        acc.x = fmaf(a, v.x, acc.x);
        acc.y = fmaf(a, v.y, acc.y);
        acc.z = fmaf(a, v.z, acc.z);
        acc.w = fmaf(a, v.w, acc.w);
    }
    const float4 iv = *(const float4*)(inp + (size_t)b * HH + h4);
    float4 r;
    r.x = iv.x + acc.x; r.y = iv.y + acc.y; r.z = iv.z + acc.z; r.w = iv.w + acc.w;
    *(float4*)(out + (size_t)b * HH + h4) = r;
}

// ---------------------------------------------------------------------------
extern "C" void kernel_launch(void* const* d_in, const int* in_sizes, int n_in,
                              void* d_out, int out_size, void* d_ws, size_t ws_size,
                              hipStream_t stream) {
    (void)in_sizes; (void)n_in; (void)out_size; (void)ws_size;

    const float* input = (const float*)d_in[0];   // [B, H]
    const float* KB    = (const float*)d_in[1];   // [K, B, H]
    const float* W     = (const float*)d_in[2];   // [H, H]
    const float* bias  = (const float*)d_in[3];   // [H]
    float* out = (float*)d_out;

    char* ws = (char*)d_ws;
    bf16_t* Wb   = (bf16_t*)ws;                           // 8 MB  : W in bf16, row-major [o][h]
    float*  q    = (float*)(ws + 8388608);                // 512 KB: q [B,H] f32
    float*  dots = (float*)(ws + 8912896);                // 64 KB : dots[R]
    float*  kn2  = (float*)(ws + 8978432);                // 64 KB : ||kproj||^2 [R]
    float*  attn = (float*)(ws + 9043968);                // 64 KB : attn[R]

    // Zero the atomic accumulators (dots & kn2 are contiguous).
    hipMemsetAsync(dots, 0, 2 * RR * sizeof(float), stream);

    // 1. W -> bf16
    cvt_bf16_kernel<<<(HH * HH / 8) / 256, 256, 0, stream>>>(W, Wb);

    // 2. q = input @ W^T + b   (WMMA, grid 1 x 16)
    proj_kernel<1><<<dim3(1, HH / 128), 256, 0, stream>>>(input, Wb, bias, nullptr,
                                                          q, nullptr, nullptr);

    // 3. Big fused GEMM: kproj norms + dots, no kproj materialization (grid 256 x 16)
    proj_kernel<0><<<dim3(RR / 64, HH / 128), 256, 0, stream>>>(KB, Wb, bias, q,
                                                                nullptr, kn2, dots);

    // 4. Cosine scores + softmax over K per b
    softmax_kernel<<<BB, KK, 0, stream>>>(q, dots, kn2, attn);

    // 5. out = input + attn-weighted sum of raw knowledge items
    output_kernel<<<dim3(HH / 1024, BB), 256, 0, stream>>>(input, KB, attn, out);
}
